// GATEncoder_48266842472717
// MI455X (gfx1250) — matrix operand order repaired and checked by reading, not compiled, not run on hardware
//
#include <hip/hip_runtime.h>
#include <hip/hip_bf16.h>

// ---------------------------------------------------------------------------
// GAT encoder for MI455X (gfx1250, wave32).
// Dense node GEMMs run on v_wmma_f32_16x16x32_bf16 (each wave computes a
// 16x64 output strip: 4 accumulators reuse one A fragment -> 4 WMMA per
// barrier pair, A read from HBM once per GEMM).  Edge softmax/scatter runs
// as bandwidth-bound atomic kernels.  All scratch carved from d_ws; all
// launches on `stream` (graph-capture safe).
// ---------------------------------------------------------------------------

typedef __attribute__((ext_vector_type(16))) __bf16 v16bf;
typedef __attribute__((ext_vector_type(8)))  float  v8f;

__device__ __forceinline__ unsigned short f2bf(float f) {
    unsigned u = __float_as_uint(f);
    unsigned r = u + 0x7FFFu + ((u >> 16) & 1u);   // round-to-nearest-even
    return (unsigned short)(r >> 16);
}

__device__ __forceinline__ void atomicMaxF(float* addr, float v) {
    // classic signed/unsigned trick; addr initialized to -inf
    if (v >= 0.0f) atomicMax((int*)addr, __float_as_int(v));
    else           atomicMin((unsigned int*)addr, __float_as_uint(v));
}

// -------------------------------- utility ----------------------------------

__global__ void k_fill_f32(float* p, float v, int n) {
    int i = blockIdx.x * blockDim.x + threadIdx.x;
    if (i < n) p[i] = v;
}

__global__ void k_f32_to_bf16(const float* __restrict__ src,
                              unsigned short* __restrict__ dst, int n) {
    int i = blockIdx.x * blockDim.x + threadIdx.x;
    if (i < n) dst[i] = f2bf(src[i]);
}

// Pack W[din,dout] (fp32, row-major) into WMMA B-fragment order (bf16):
// chunk index = (ntile * nK + kt), 32 lanes x 16 bf16 per chunk.
// lane<16 : col = ntile*16+lane,    K = kt*32 + j       (j=0..15)
// lane>=16: col = ntile*16+lane-16, K = kt*32 + 16 + j
__global__ void k_pack_w(const float* __restrict__ W,
                         unsigned short* __restrict__ Bp,
                         int din, int dout) {
    int idx = blockIdx.x * blockDim.x + threadIdx.x;
    int nK = din >> 5;
    int total = (dout >> 4) * nK * 32;
    if (idx >= total) return;
    int lane  = idx & 31;
    int kt    = (idx >> 5) % nK;
    int ntile = idx / (32 * nK);
    int col   = ntile * 16 + (lane & 15);
    int kbase = kt * 32 + ((lane < 16) ? 0 : 16);
    unsigned short* dst = Bp + (size_t)idx * 16;
#pragma unroll
    for (int j = 0; j < 16; ++j)
        dst[j] = f2bf(W[(size_t)(kbase + j) * dout + col]);
}

// ------------------------------ WMMA GEMM ----------------------------------
// H[N,dout] = A[N,din](bf16) @ Bp(packed bf16).
// One block = one 16-row M tile covering ALL dout columns.
// 4 waves; each wave owns T adjacent 16-col N tiles (T = dout/64):
// one A fragment (from LDS) feeds T WMMAs per K-step.
template<int T>
__global__ __launch_bounds__(128)
void k_gemm_bf16_wmma(const unsigned short* __restrict__ A,
                      const unsigned short* __restrict__ Bp,
                      float* __restrict__ Hout,
                      int Nrows, int din, int dout) {
    const int mtile = blockIdx.x;
    const int wave  = threadIdx.x >> 5;
    const int lane  = threadIdx.x & 31;
    const int nK    = din >> 5;
    const int nt0   = wave * T;               // first N-tile of this wave

    __shared__ unsigned short As[16 * 32];    // 16 rows x 32 bf16 = 1 KB
    unsigned int* Asu = (unsigned int*)As;

    v8f acc[T];
#pragma unroll
    for (int t = 0; t < T; ++t) acc[t] = (v8f){};

    for (int kt = 0; kt < nK; ++kt) {
        const int k = kt << 5;
        __syncthreads();
        // cooperative global -> LDS stage of the A tile (256 dwords)
#pragma unroll
        for (int i = 0; i < 2; ++i) {
            int u   = threadIdx.x + i * 128;
            int row = u >> 4, kp = u & 15;
            int grow = mtile * 16 + row;
            if (grow >= Nrows) grow = Nrows - 1;
            const unsigned int* src =
                (const unsigned int*)(A + (size_t)grow * din + k);
            Asu[u] = src[kp];
        }
        __syncthreads();

        // A fragment from LDS (row = lane&15, k = 16*(lane/16) + s, same k order as Bp)
        const int arow = lane & 15;
        const int koff = (lane < 16) ? 0 : 16;
        union { uint4 q[2]; v16bf v; } af;
        af.q[0] = *(const uint4*)(As + arow * 32 + koff);
        af.q[1] = *(const uint4*)(As + arow * 32 + koff + 8);

        // T B-fragments: contiguous 32 B per lane from packed weights
#pragma unroll
        for (int t = 0; t < T; ++t) {
            const unsigned short* bsrc =
                Bp + (((size_t)(nt0 + t) * nK + kt) * 32 + lane) * 16;
            union { uint4 q[2]; v16bf v; } bfR;
            bfR.q[0] = ((const uint4*)bsrc)[0];
            bfR.q[1] = ((const uint4*)bsrc)[1];
            // prefetch next K-step of this tile's weights (global_prefetch_b8)
            if (kt + 1 < nK)
                __builtin_prefetch((const void*)(bsrc + 32 * 16), 0, 1);
            acc[t] = __builtin_amdgcn_wmma_f32_16x16x32_bf16(
                false, af.v, false, bfR.v, (short)0, acc[t], false, false);
        }
    }

    // C/D layout: lane<16 -> rows 0..7, lane>=16 -> rows 8..15; col = lane&15
    const int rbase = mtile * 16 + ((lane < 16) ? 0 : 8);
#pragma unroll
    for (int t = 0; t < T; ++t) {
        const int col = (nt0 + t) * 16 + (lane & 15);
#pragma unroll
        for (int r = 0; r < 8; ++r) {
            int row = rbase + r;
            if (row < Nrows) Hout[(size_t)row * dout + col] = acc[t][r];
        }
    }
}

// ------------------------------ attention ----------------------------------

__global__ void k_alpha(const float* __restrict__ H,
                        const float* __restrict__ a_s,
                        const float* __restrict__ a_d,
                        float* __restrict__ as_out,
                        float* __restrict__ ad_out,
                        int Nn, int heads, int ch) {
    int idx = blockIdx.x * blockDim.x + threadIdx.x;
    if (idx >= Nn * heads) return;
    int n = idx / heads, hd = idx % heads;
    const float* hp = H + (size_t)n * heads * ch + (size_t)hd * ch;
    float s = 0.f, d = 0.f;
    for (int c = 0; c < ch; ++c) {
        float v = hp[c];
        s += v * a_s[hd * ch + c];
        d += v * a_d[hd * ch + c];
    }
    as_out[idx] = s;
    ad_out[idx] = d;
}

__global__ void k_edge_logits_max(const int* __restrict__ srcI,
                                  const int* __restrict__ dstI,
                                  const float* __restrict__ as_,
                                  const float* __restrict__ ad_,
                                  float* __restrict__ lg,
                                  float* __restrict__ mx,
                                  int E, int Etot, int heads) {
    int idx = blockIdx.x * blockDim.x + threadIdx.x;
    if (idx >= Etot * heads) return;
    int e = idx / heads, hd = idx % heads;
    int s = (e < E) ? srcI[e] : (e - E);
    int d = (e < E) ? dstI[e] : (e - E);
    float v = as_[s * heads + hd] + ad_[d * heads + hd];
    v = (v > 0.f) ? v : 0.2f * v;                  // leaky_relu(0.2)
    lg[idx] = v;
    atomicMaxF(&mx[d * heads + hd], v);
}

__global__ void k_edge_exp_den(const int* __restrict__ srcI,
                               const int* __restrict__ dstI,
                               float* __restrict__ lg,     // in: logit, out: exp
                               const float* __restrict__ mx,
                               float* __restrict__ den,
                               int E, int Etot, int heads) {
    int idx = blockIdx.x * blockDim.x + threadIdx.x;
    if (idx >= Etot * heads) return;
    int e = idx / heads, hd = idx % heads;
    int d = (e < E) ? dstI[e] : (e - E);
    float ex = __expf(lg[idx] - mx[d * heads + hd]);
    lg[idx] = ex;
    atomicAdd(&den[d * heads + hd], ex);
}

// one wave per edge: scatter h[src] * alpha into out[dst]
__global__ __launch_bounds__(256)
void k_edge_scatter(const int* __restrict__ srcI,
                    const int* __restrict__ dstI,
                    const float* __restrict__ H,
                    const float* __restrict__ ex,
                    const float* __restrict__ den,
                    float* __restrict__ Out,
                    int E, int Etot, int heads, int ch) {
    int e = blockIdx.x * 8 + (threadIdx.x >> 5);
    if (e >= Etot) return;
    int lane = threadIdx.x & 31;
    int s = (e < E) ? srcI[e] : (e - E);
    int d = (e < E) ? dstI[e] : (e - E);
    int dim = heads * ch;
    const float* hs = H + (size_t)s * dim;
    float* od = Out + (size_t)d * dim;
    for (int c = lane; c < dim; c += 32) {
        int hd = c / ch;
        float w = ex[(size_t)e * heads + hd] /
                  (den[d * heads + hd] + 1e-16f);
        atomicAdd(&od[c], hs[c] * w);
    }
}

// BN(eval) + ELU, in place; also emit bf16 copy for the next layer's GEMM
__global__ void k_bn_elu(float* __restrict__ Out,
                         const float* __restrict__ bias,
                         const float* __restrict__ gamma,
                         const float* __restrict__ beta,
                         const float* __restrict__ mean,
                         const float* __restrict__ var,
                         unsigned short* __restrict__ Xb,
                         int Nn, int dout) {
    int idx = blockIdx.x * blockDim.x + threadIdx.x;
    if (idx >= Nn * dout) return;
    int d = idx % dout;
    float v = Out[idx] + bias[d];
    float y = (v - mean[d]) * __frsqrt_rn(var[d] + 1e-5f) * gamma[d] + beta[d];
    float r = (y > 0.f) ? y : (__expf(y) - 1.0f);   // ELU
    Out[idx] = r;
    Xb[idx]  = f2bf(r);
}

// ------------------------------- pooling -----------------------------------

__global__ void k_pool_accum(const float* __restrict__ act,
                             const int* __restrict__ batch,
                             float* __restrict__ sum, int Nn) {
    int idx = blockIdx.x * blockDim.x + threadIdx.x;
    if (idx >= Nn * 64) return;
    int n = idx >> 6, d = idx & 63;
    atomicAdd(&sum[batch[n] * 64 + d], act[(size_t)n * 64 + d]);
}

__global__ void k_pool_cnt(const int* __restrict__ batch,
                           float* __restrict__ cnt, int Nn) {
    int n = blockIdx.x * blockDim.x + threadIdx.x;
    if (n < Nn) atomicAdd(&cnt[batch[n]], 1.0f);
}

__global__ void k_final_emb(const float* __restrict__ sum,
                            const float* __restrict__ cnt,
                            const float* __restrict__ Wemb,
                            const float* __restrict__ bemb,
                            float* __restrict__ out) {
    int idx = blockIdx.x * blockDim.x + threadIdx.x;   // 32*64
    if (idx >= 32 * 64) return;
    int b = idx >> 6, e = idx & 63;
    float cr = 1.0f / fmaxf(cnt[b], 1.0f);
    float acc = 0.f;
    for (int c = 0; c < 64; ++c)
        acc += sum[b * 64 + c] * cr * Wemb[c * 64 + e];
    out[idx] = acc + bemb[e];
}

// ------------------------------- launcher ----------------------------------

extern "C" void kernel_launch(void* const* d_in, const int* in_sizes, int n_in,
                              void* d_out, int out_size, void* d_ws, size_t ws_size,
                              hipStream_t stream) {
    const float* x    = (const float*)d_in[0];
    const int*   ei   = (const int*)d_in[1];
    const int*   batch= (const int*)d_in[2];

    const int N    = in_sizes[2];          // 20000
    const int E    = in_sizes[1] / 2;      // 320000
    const int Etot = E + N;                // + self loops
    const int H    = 4, C = 64;
    const int*   srcI = ei;
    const int*   dstI = ei + E;

    // per-layer parameter pointers (dict order: W,asrc,adst,bias,gamma,beta,mean,var)
    const float *W[3], *as_[3], *ad_[3], *bias[3], *gam[3], *bet[3], *mean[3], *var[3];
    for (int l = 0; l < 3; ++l) {
        int b0 = 3 + l * 8;
        W[l]    = (const float*)d_in[b0 + 0];
        as_[l]  = (const float*)d_in[b0 + 1];
        ad_[l]  = (const float*)d_in[b0 + 2];
        bias[l] = (const float*)d_in[b0 + 3];
        gam[l]  = (const float*)d_in[b0 + 4];
        bet[l]  = (const float*)d_in[b0 + 5];
        mean[l] = (const float*)d_in[b0 + 6];
        var[l]  = (const float*)d_in[b0 + 7];
    }
    const float* Wemb = (const float*)d_in[27];
    const float* bemb = (const float*)d_in[28];
    float* out = (float*)d_out;

    // ---- workspace partition (256-B aligned) ----
    char* ws = (char*)d_ws;
    size_t off = 0;
    auto alloc = [&](size_t bytes) -> char* {
        char* p = ws + off;
        off = (off + bytes + 255) & ~(size_t)255;
        return p;
    };
    unsigned short* xb  = (unsigned short*)alloc((size_t)N * 256 * 2); // bf16 acts
    float* hbuf         = (float*)alloc((size_t)N * 256 * 4);          // x@W
    float* obuf         = (float*)alloc((size_t)N * 256 * 4);          // aggregated
    float* lg           = (float*)alloc((size_t)Etot * H * 4);         // logits/exp
    float* asN          = (float*)alloc((size_t)N * H * 4);
    float* adN          = (float*)alloc((size_t)N * H * 4);
    float* mx           = (float*)alloc((size_t)N * H * 4);
    float* den          = (float*)alloc((size_t)N * H * 4);
    unsigned short* Wp  = (unsigned short*)alloc((size_t)256 * 256 * 2);
    float* psum         = (float*)alloc(32 * 64 * 4);
    float* pcnt         = (float*)alloc(32 * 4);
    (void)ws_size; (void)n_in; (void)out_size;

    const int TB = 256;
    auto cdiv = [](int a, int b) { return (a + b - 1) / b; };

    // layer configs: (din, heads)
    const int dinL[3]   = {128, 256, 256};
    const int headsL[3] = {H, H, 1};

    // input -> bf16
    k_f32_to_bf16<<<cdiv(N * 128, TB), TB, 0, stream>>>(x, xb, N * 128);

    for (int l = 0; l < 3; ++l) {
        const int din   = dinL[l];
        const int heads = headsL[l];
        const int dout  = heads * C;      // 256,256,64 (concat / single head)
        const int nK    = din >> 5;

        // pack weights into B-fragment order
        int packN = (dout >> 4) * nK * 32;
        k_pack_w<<<cdiv(packN, TB), TB, 0, stream>>>(W[l], Wp, din, dout);

        // h = xb @ W  (WMMA bf16): one block per 16 rows, full dout width
        if (dout == 256)
            k_gemm_bf16_wmma<4><<<cdiv(N, 16), 128, 0, stream>>>(
                xb, Wp, hbuf, N, din, dout);
        else
            k_gemm_bf16_wmma<1><<<cdiv(N, 16), 128, 0, stream>>>(
                xb, Wp, hbuf, N, din, dout);

        // per-node attention coefficients
        k_alpha<<<cdiv(N * heads, TB), TB, 0, stream>>>(
            hbuf, as_[l], ad_[l], asN, adN, N, heads, C);

        // init segment-softmax state + output accumulator
        k_fill_f32<<<cdiv(N * heads, TB), TB, 0, stream>>>(mx, -INFINITY, N * heads);
        k_fill_f32<<<cdiv(N * heads, TB), TB, 0, stream>>>(den, 0.f, N * heads);
        k_fill_f32<<<cdiv(N * dout, TB), TB, 0, stream>>>(obuf, 0.f, N * dout);

        // edge phase: max -> exp/den -> weighted scatter
        k_edge_logits_max<<<cdiv(Etot * heads, TB), TB, 0, stream>>>(
            srcI, dstI, asN, adN, lg, mx, E, Etot, heads);
        k_edge_exp_den<<<cdiv(Etot * heads, TB), TB, 0, stream>>>(
            srcI, dstI, lg, mx, den, E, Etot, heads);
        k_edge_scatter<<<cdiv(Etot, 8), 256, 0, stream>>>(
            srcI, dstI, hbuf, lg, den, obuf, E, Etot, heads, C);

        // bias + BN + ELU (in place) and bf16 copy for next GEMM
        k_bn_elu<<<cdiv(N * dout, TB), TB, 0, stream>>>(
            obuf, bias[l], gam[l], bet[l], mean[l], var[l], xb, N, dout);
    }

    // global mean pool + final embedding
    k_fill_f32<<<cdiv(32 * 64, TB), TB, 0, stream>>>(psum, 0.f, 32 * 64);
    k_fill_f32<<<1, 64, 0, stream>>>(pcnt, 0.f, 32);
    k_pool_accum<<<cdiv(N * 64, TB), TB, 0, stream>>>(obuf, batch, psum, N);
    k_pool_cnt<<<cdiv(N, TB), TB, 0, stream>>>(batch, pcnt, N);
    k_final_emb<<<cdiv(32 * 64, TB), TB, 0, stream>>>(psum, pcnt, Wemb, bemb, out);
}
